// Angel_Unit_61048665145726
// MI455X (gfx1250) — matrix operand order repaired and checked by reading, not compiled, not run on hardware
//
#include <hip/hip_runtime.h>
#include <hip/hip_bf16.h>

typedef __attribute__((ext_vector_type(16))) _Float16 v16h;
typedef __attribute__((ext_vector_type(8)))  float    v8f;

__device__ __forceinline__ float sigmoidf_(float x) { return 1.0f / (1.0f + __expf(-x)); }

__device__ __forceinline__ v8f wmma_f16(v16h a, v16h b, v8f c) {
    // D = A(16x32 f16) * B(32x16 f16) + C(16x16 f32)
    return __builtin_amdgcn_wmma_f32_16x16x32_f16(false, a, false, b, (short)0, c, false, false);
}

// ---------------------------------------------------------------------------
// Conv1d(128->128, k=5, VALID) as implicit GEMM.
// One block = one (posTile, batch): 8 waves = 8 co-tiles share one LDS-staged
// input patch x[b, 0:128, p0:p0+20] (f16, 5KB, loaded exactly once; staging is
// division-free: thread t owns row t>>1, half t&1, 10 unrolled columns).
// K = ci*5 + k = 640 -> 20 WMMA K-steps; B fragments gathered from LDS.
// Output stored f16 in [b][co*1020+p] flat order so the torch ".view"
// reinterpretation seq[b][t][d] = flat[t*128+d] is pure indexing.
// ---------------------------------------------------------------------------
__global__ __launch_bounds__(256)
void conv_wmma(const float* __restrict__ x, const float* __restrict__ w,
               const float* __restrict__ bias, _Float16* __restrict__ seq)
{
    __shared__ _Float16 xl[128][20];

    const int tid  = threadIdx.x;
    const int coT  = tid >> 5;          // wave id = co tile
    const int lane = tid & 31;
    const int g  = lane >> 4;
    const int ln = lane & 15;
    const int pT = blockIdx.x, b = blockIdx.y;
    const int p0 = pT * 16;
    const float* xb = x + (size_t)b * (128 * 1024);

    // stage input patch once, f32 -> f16 (zero-fill past the input edge)
    {
        const int ci   = tid >> 1;           // 0..127
        const int pp0  = (tid & 1) * 10;     // 0 or 10
        const float* xr = xb + ci * 1024 + p0 + pp0;
#pragma unroll
        for (int k = 0; k < 10; ++k) {
            int pos = p0 + pp0 + k;
            xl[ci][pp0 + k] = (pos < 1024) ? (_Float16)xr[k] : (_Float16)0.0f;
        }
    }
    __syncthreads();

    v8f acc;
#pragma unroll
    for (int v = 0; v < 8; ++v) acc[v] = bias[coT * 16 + v + 8 * g];  // D: M = v+8g

    for (int kb = 0; kb < 20; ++kb) {
        // A fragment: weights, contiguous K pairs (L2-resident)
        v16h A;
#pragma unroll
        for (int v = 0; v < 8; ++v) {
            int K = kb * 32 + ((v < 4) ? 0 : 16) + g * 8 + (v & 3) * 2;
            const float* wr = w + (size_t)(coT * 16 + ln) * 640 + K;
            A[2 * v]     = (_Float16)wr[0];
            A[2 * v + 1] = (_Float16)wr[1];
        }
        // B fragment: input patch from LDS, B[K][n] = x[ci][n+kr], K=ci*5+kr
        v16h Bm;
#pragma unroll
        for (int v = 0; v < 8; ++v) {
#pragma unroll
            for (int hh = 0; hh < 2; ++hh) {
                int kz = kb * 32 + ((v < 4) ? 0 : 16) + g * 8 + (v & 3) * 2 + hh;
                int ci = kz / 5, kr = kz - ci * 5;   // compile-time per (kb,v,hh) once unrolled
                Bm[2 * v + hh] = xl[ci][ln + kr];
            }
        }
        acc = wmma_f16(A, Bm, acc);
    }

    const int p = p0 + ln;
    if (p < 1020) {
        _Float16* sb = seq + (size_t)b * (128 * 1020);
#pragma unroll
        for (int v = 0; v < 8; ++v) {
            int co = coT * 16 + v + 8 * g;
            sb[co * 1020 + p] = (_Float16)acc[v];
        }
    }
}

// ---------------------------------------------------------------------------
// Input-gate precompute GEMM: out[row][gate] = in[row][:IND] @ Wih^T + bih+bhh
// rows = B*T = 522240 (batch*time parallel). K zero-padded to 32 when IND=16.
// Each block processes MREP row tiles so the Wih^T B-fragments and bias are
// loaded once and reused MREP times from registers.
// ---------------------------------------------------------------------------
template<int IND, int GH, int MREP>
__global__ __launch_bounds__(32)
void xg_wmma(const _Float16* __restrict__ in, const float* __restrict__ Wih,
             const float* __restrict__ bih, const float* __restrict__ bhh,
             _Float16* __restrict__ out)
{
    const int lane = threadIdx.x;
    const int g  = lane >> 4;
    const int ln = lane & 15;
    const int r0 = blockIdx.x * (16 * MREP);
    const int nt = blockIdx.y;
    const int gate = nt * 16 + ln;

    constexpr int KSTEPS = (IND + 31) / 32;

    const float bb = bih[gate] + bhh[gate];

    // Weight B fragments, loaded once per block
    v16h Bk[KSTEPS];
#pragma unroll
    for (int kb = 0; kb < KSTEPS; ++kb) {
#pragma unroll
        for (int v = 0; v < 8; ++v) {
#pragma unroll
            for (int hh = 0; hh < 2; ++hh) {
                int K = kb * 32 + ((v < 4) ? 0 : 16) + g * 8 + (v & 3) * 2 + hh;
                Bk[kb][2 * v + hh] = (K < IND) ? (_Float16)Wih[gate * IND + K] : (_Float16)0.0f;
            }
        }
    }

    for (int mr = 0; mr < MREP; ++mr) {
        const int rt = r0 + mr * 16;
        v8f acc;
#pragma unroll
        for (int v = 0; v < 8; ++v) acc[v] = bb;   // bias depends only on N=gate

#pragma unroll
        for (int kb = 0; kb < KSTEPS; ++kb) {
            v16h A;
            const _Float16* ir = in + (size_t)(rt + ln) * IND;
#pragma unroll
            for (int v = 0; v < 8; ++v) {
                int K = kb * 32 + ((v < 4) ? 0 : 16) + g * 8 + (v & 3) * 2;
                A[2 * v]     = (K     < IND) ? ir[K]     : (_Float16)0.0f;
                A[2 * v + 1] = (K + 1 < IND) ? ir[K + 1] : (_Float16)0.0f;
            }
            acc = wmma_f16(A, Bk[kb], acc);
        }
#pragma unroll
        for (int v = 0; v < 8; ++v) {
            int row = rt + v + 8 * g;
            out[(size_t)row * GH + gate] = (_Float16)acc[v];
        }
    }
}

// ---------------------------------------------------------------------------
// LSTM recurrence: one workgroup (4 waves) owns 16 batches for all 1020 steps.
// Per step: each wave computes NT/4 gate tiles of h @ Whh^T via WMMA (Whh^T
// B-fragments VGPR-resident for the whole scan, C seeded zero), results land
// in LDS; the elementwise gate update (+ precomputed xg, loaded coalesced)
// is spread over all 128 threads. h kept f16 row-major, K-padded to 32.
// ---------------------------------------------------------------------------
template<int H, bool WRITE_H, bool WRITE_LAST>
__global__ __launch_bounds__(128)
void lstm_rec(const _Float16* __restrict__ xg, const float* __restrict__ Whh,
              _Float16* __restrict__ hout, float* __restrict__ last)
{
    constexpr int GH  = 4 * H;
    constexpr int NT  = GH / 16;
    constexpr int TPW = NT / 4;       // gate tiles per wave (2 for H=32, 1 for H=16)
    __shared__ float    Gs[16][GH];   // recurrent gate contributions, f32
    __shared__ float    Cs[16][H];    // cell state
    __shared__ _Float16 Hs[16][32];   // hidden state, K-padded to 32

    const int tid  = threadIdx.x;
    const int wv   = tid >> 5;
    const int lane = tid & 31;
    const int g  = lane >> 4;
    const int ln = lane & 15;
    const int b0 = blockIdx.x * 16;

    // Recurrent-weight B fragments for this wave's tiles, VGPR-resident
    v16h Bw[TPW];
#pragma unroll
    for (int q = 0; q < TPW; ++q) {
        int nt = wv * TPW + q;
#pragma unroll
        for (int v = 0; v < 8; ++v) {
#pragma unroll
            for (int hh = 0; hh < 2; ++hh) {
                int K = ((v < 4) ? 0 : 16) + g * 8 + (v & 3) * 2 + hh;
                int gate = nt * 16 + ln;
                Bw[q][2 * v + hh] = (K < H) ? (_Float16)Whh[gate * H + K] : (_Float16)0.0f;
            }
        }
    }
    for (int i = tid; i < 16 * 32; i += 128) Hs[i >> 5][i & 31] = (_Float16)0.0f;
    for (int i = tid; i < 16 * H; i += 128) Cs[i / H][i % H] = 0.0f;
    __syncthreads();

    for (int t = 0; t < 1020; ++t) {
        // A fragment: h tile, contiguous K pairs from LDS (same in every wave)
        v16h A;
#pragma unroll
        for (int v = 0; v < 8; ++v) {
            int K = ((v < 4) ? 0 : 16) + g * 8 + (v & 3) * 2;
            A[2 * v]     = Hs[ln][K];
            A[2 * v + 1] = Hs[ln][K + 1];
        }
#pragma unroll
        for (int q = 0; q < TPW; ++q) {
            int nt = wv * TPW + q;
            v8f acc = {};                       // C = 0
            acc = wmma_f16(A, Bw[q], acc);
#pragma unroll
            for (int v = 0; v < 8; ++v) Gs[v + 8 * g][nt * 16 + ln] = acc[v];
        }
        __syncthreads();
        for (int i = tid; i < 16 * H; i += 128) {
            int bl = i / H, j = i % H;
            size_t rbase = ((size_t)(b0 + bl) * 1020 + t) * GH;
            float iv = Gs[bl][j]         + (float)xg[rbase + j];
            float fv = Gs[bl][H + j]     + (float)xg[rbase + H + j];
            float gv = Gs[bl][2 * H + j] + (float)xg[rbase + 2 * H + j];
            float ov = Gs[bl][3 * H + j] + (float)xg[rbase + 3 * H + j];
            float c  = Cs[bl][j];
            c = sigmoidf_(fv) * c + sigmoidf_(iv) * tanhf(gv);
            float hv = sigmoidf_(ov) * tanhf(c);
            Cs[bl][j] = c;
            Hs[bl][j] = (_Float16)hv;
            if constexpr (WRITE_H)
                hout[((size_t)(b0 + bl) * 1020 + t) * H + j] = (_Float16)hv;
            if constexpr (WRITE_LAST)
                if (t == 1019) last[(b0 + bl) * H + j] = hv;
        }
        __syncthreads();
    }
}

// ---------------------------------------------------------------------------
// Workspace layout (bytes), ~268 MB peak, regions reused after producer dies:
//   seq  @ 0          : 512*1020*128*2 = 133693440   (dead after xg1)
//   xg1  @ 133693440  : 133693440                    (dead after rec1)
//   h1   @ 0          : 512*1020*32*2  =  33423360
//   xg2  @ 33423360   : 512*1020*64*2  =  66846720
//   h2   @ 100270080  : 512*1020*16*2  =  16711680
//   xg3  @ 116981760  : 133693440      -> end 250675200
// ---------------------------------------------------------------------------
extern "C" void kernel_launch(void* const* d_in, const int* in_sizes, int n_in,
                              void* d_out, int out_size, void* d_ws, size_t ws_size,
                              hipStream_t stream)
{
    const float* x     = (const float*)d_in[0];
    const float* convw = (const float*)d_in[1];
    const float* convb = (const float*)d_in[2];
    const float* Wih1  = (const float*)d_in[3];
    const float* Whh1  = (const float*)d_in[4];
    const float* bih1  = (const float*)d_in[5];
    const float* bhh1  = (const float*)d_in[6];
    const float* Wih2  = (const float*)d_in[7];
    const float* Whh2  = (const float*)d_in[8];
    const float* bih2  = (const float*)d_in[9];
    const float* bhh2  = (const float*)d_in[10];
    const float* Wih3  = (const float*)d_in[11];
    const float* Whh3  = (const float*)d_in[12];
    const float* bih3  = (const float*)d_in[13];
    const float* bhh3  = (const float*)d_in[14];
    float* out = (float*)d_out;

    char* ws = (char*)d_ws;
    _Float16* seq = (_Float16*)(ws + 0);
    _Float16* xg1 = (_Float16*)(ws + 133693440u);
    _Float16* h1  = (_Float16*)(ws + 0);
    _Float16* xg2 = (_Float16*)(ws + 33423360u);
    _Float16* h2  = (_Float16*)(ws + 100270080u);
    _Float16* xg3 = (_Float16*)(ws + 116981760u);

    // 1) conv -> seq (f16): one block per (posTile, batch), 8 waves = co tiles
    conv_wmma<<<dim3(64, 512), 256, 0, stream>>>(x, convw, convb, seq);
    // 2) layer 1: gates GEMM then recurrence (rows/128 = 4080 blocks)
    xg_wmma<128, 128, 8><<<dim3(4080, 8), 32, 0, stream>>>(seq, Wih1, bih1, bhh1, xg1);
    lstm_rec<32, true, false><<<32, 128, 0, stream>>>(xg1, Whh1, h1, nullptr);
    // 3) layer 2
    xg_wmma<32, 64, 8><<<dim3(4080, 4), 32, 0, stream>>>(h1, Wih2, bih2, bhh2, xg2);
    lstm_rec<16, true, false><<<32, 128, 0, stream>>>(xg2, Whh2, h2, nullptr);
    // 4) layer 3: only the final hidden state is needed
    xg_wmma<16, 128, 8><<<dim3(4080, 8), 32, 0, stream>>>(h2, Wih3, bih3, bhh3, xg3);
    lstm_rec<32, false, true><<<32, 128, 0, stream>>>(xg3, Whh3, nullptr, out);
}
